// W8A8Linear_77610059039048
// MI455X (gfx1250) — 8-bit matrix operand order repaired and checked by reading, compiled, not run on hardware
//
#include <hip/hip_runtime.h>
#include <stdint.h>

typedef int      v8i __attribute__((ext_vector_type(8)));
typedef int      v4i __attribute__((ext_vector_type(4)));
typedef _Float16 v8h __attribute__((ext_vector_type(8)));
typedef _Float16 v4h __attribute__((ext_vector_type(4)));

#define QMAXF 127.0f
#define EPSF  1e-5f

union V8U { v8i v; v4i q[2]; };

__global__ __launch_bounds__(256) void w8a8_rowmax(const _Float16* __restrict__ x,
                                                   float* __restrict__ scales,
                                                   int K) {
    __shared__ float red[256];
    const int row = blockIdx.x;
    const int t = threadIdx.x;
    const _Float16* xr = x + (size_t)row * K;

    float m = 0.0f;
    for (int k = t * 8; k < K; k += 256 * 8) {
        v8h v = *(const v8h*)(xr + k);
#pragma unroll
        for (int i = 0; i < 8; ++i) {
            float f = fabsf((float)v[i]);
            m = fmaxf(m, f);
        }
    }
    red[t] = m;
    __syncthreads();
    for (int s = 128; s > 0; s >>= 1) {
        if (t < s) red[t] = fmaxf(red[t], red[t + s]);
        __syncthreads();
    }
    if (t == 0) scales[row] = fmaxf(red[0], EPSF) / QMAXF;
}

__global__ __launch_bounds__(256) void w8a8_quantpack(const _Float16* __restrict__ x,
                                                      const float* __restrict__ scales,
                                                      int8_t* __restrict__ apk,
                                                      int K) {
    const int lane = threadIdx.x & 31;
    const int waveInBlk = threadIdx.x >> 5;
    const int Ktiles = K >> 6;

    const int tile  = blockIdx.x * 8 + waveInBlk;
    const int mtile = tile / Ktiles;
    const int ktile = tile % Ktiles;

    const int r   = lane & 15;
    const int h   = lane >> 4;
    const int row = mtile * 16 + r;

    const float inv = 1.0f / scales[row];
    const _Float16* xr = x + (size_t)row * K + ktile * 64;

    int d[8];
#pragma unroll
    for (int j = 0; j < 8; ++j) {
        const int kb = ((j >> 2) << 5) + (h << 4) + ((j & 3) << 2);
        v4h v = *(const v4h*)(xr + kb);
        int q0 = (int)__builtin_rintf((float)v[0] * inv);
        int q1 = (int)__builtin_rintf((float)v[1] * inv);
        int q2 = (int)__builtin_rintf((float)v[2] * inv);
        int q3 = (int)__builtin_rintf((float)v[3] * inv);
        d[j] = (q0 & 255) | ((q1 & 255) << 8) | ((q2 & 255) << 16) | ((q3 & 255) << 24);
    }
    v4i lo = { d[0], d[1], d[2], d[3] };
    v4i hi = { d[4], d[5], d[6], d[7] };
    v4i* dst = (v4i*)(apk + ((size_t)tile << 10) + lane * 32);
    dst[0] = lo;
    dst[1] = hi;
}

__global__ __launch_bounds__(256) void w8a8_gemm(const int8_t* __restrict__ apk,
                                                 const int8_t* __restrict__ w,
                                                 const float* __restrict__ scaleA,
                                                 const float* __restrict__ scaleW,
                                                 const _Float16* __restrict__ bias,
                                                 _Float16* __restrict__ out,
                                                 int N, int K) {
    const int lane = threadIdx.x & 31;
    const int wave = threadIdx.x >> 5;
    const int wm = wave & 3;
    const int wn = wave >> 2;
    const int Ktiles = K >> 6;

    const int mBase = blockIdx.y * 256 + wm * 64;
    const int nBase = blockIdx.x * 128 + wn * 64;

    const int h  = lane >> 4;
    const int nl = lane & 15;

    v8i acc[4][4];
#pragma unroll
    for (int mt = 0; mt < 4; ++mt)
#pragma unroll
        for (int nt = 0; nt < 4; ++nt)
            acc[mt][nt] = (v8i){0, 0, 0, 0, 0, 0, 0, 0};

    const char* ab = (const char*)apk + (((size_t)(mBase >> 4) * Ktiles) << 10) + (size_t)lane * 32;
    const char* bb = (const char*)w + (size_t)(nBase + nl) * K + h * 16;
    const size_t aStride = ((size_t)Ktiles << 10);
    const size_t bStride = ((size_t)K << 4);

    for (int kt = 0; kt < Ktiles; ++kt) {
        V8U a[4];
#pragma unroll
        for (int mt = 0; mt < 4; ++mt) {
            const v4i* p = (const v4i*)(ab + (size_t)mt * aStride + ((size_t)kt << 10));
            a[mt].q[0] = p[0];
            a[mt].q[1] = p[1];
        }
#pragma unroll
        for (int nt = 0; nt < 4; ++nt) {
            const v4i* p = (const v4i*)(bb + (size_t)nt * bStride + (size_t)kt * 64);
            V8U b;
            b.q[0] = p[0];
            b.q[1] = p[2];
#pragma unroll
            for (int mt = 0; mt < 4; ++mt)
                acc[mt][nt] = __builtin_amdgcn_wmma_i32_16x16x64_iu8(
                    true, a[mt].v, true, b.v, acc[mt][nt], false, false);
        }
    }

    float sa[4][8];
#pragma unroll
    for (int mt = 0; mt < 4; ++mt)
#pragma unroll
        for (int j = 0; j < 8; ++j)
            sa[mt][j] = scaleA[mBase + mt * 16 + h * 8 + j];

#pragma unroll
    for (int nt = 0; nt < 4; ++nt) {
        const int col = nBase + nt * 16 + nl;
        const float sw = scaleW[col];
        const _Float16 bv = bias[col];
#pragma unroll
        for (int mt = 0; mt < 4; ++mt) {
#pragma unroll
            for (int j = 0; j < 8; ++j) {
                const int row = mBase + mt * 16 + h * 8 + j;
                float f = (float)acc[mt][nt][j] * sa[mt][j] * sw;
                _Float16 o = (_Float16)f + bv;
                __builtin_nontemporal_store(o, &out[(size_t)row * N + col]);
            }
        }
    }
}

extern "C" void kernel_launch(void* const* d_in, const int* in_sizes, int n_in,
                              void* d_out, int out_size, void* d_ws, size_t ws_size,
                              hipStream_t stream) {
    (void)n_in; (void)out_size; (void)ws_size;

    const _Float16* x       = (const _Float16*)d_in[0];
    const int8_t*   weight  = (const int8_t*)d_in[1];
    const float*    scale_c = (const float*)d_in[2];
    const _Float16* bias    = (const _Float16*)d_in[3];
    _Float16*       out     = (_Float16*)d_out;

    const int N = in_sizes[2];
    const int K = (int)((long long)in_sizes[1] / N);
    const int M = (int)((long long)in_sizes[0] / K);

    int8_t* apk    = (int8_t*)d_ws;
    float*  scales = (float*)((char*)d_ws + (size_t)M * K);

    w8a8_rowmax<<<M, 256, 0, stream>>>(x, scales, K);

    const int tiles = (M / 16) * (K / 64);
    w8a8_quantpack<<<tiles / 8, 256, 0, stream>>>(x, scales, apk, K);

    dim3 grid(N / 128, M / 256);
    w8a8_gemm<<<grid, 256, 0, stream>>>(apk, weight, scales, scale_c, bias, out, N, K);
}
